// MixtureOfExperts_20100446945836
// MI455X (gfx1250) — matrix operand order repaired and checked by reading, not compiled, not run on hardware
//
#include <hip/hip_runtime.h>

#define CDIM 1024
#define HDIM 4096
#define NROUTED 7
#define BM 128
#define BN 128
#define BK 32
#define LDSK 40   // padded LDS row stride in bf16 elements (80B) to spread banks

typedef __attribute__((ext_vector_type(16))) __bf16 v16bf;
typedef __attribute__((ext_vector_type(8)))  float  v8f;

struct alignas(16) U128 { unsigned x, y, z, w; };
struct alignas(8)  U64  { unsigned x, y; };
struct alignas(32) F256 { U128 a, b; };

__device__ inline unsigned short f2bf(float f) {
    unsigned u = __builtin_bit_cast(unsigned, f);
    unsigned r = (u + 0x7fffu + ((u >> 16) & 1u)) >> 16;   // RNE
    return (unsigned short)r;
}

__device__ inline float gelu_exact(float x) {
    return 0.5f * x * (1.0f + erff(x * 0.70710678118654752440f));
}

// ---------------- fp32 -> bf16 convert (x) ----------------
__global__ void k_cvt_bf16(const float* __restrict__ in, unsigned short* __restrict__ out, long nElem) {
    long i = ((long)blockIdx.x * blockDim.x + threadIdx.x) * 4;
    if (i + 3 < nElem) {
        float4 v = *(const float4*)(in + i);
        U64 o;
        o.x = (unsigned)f2bf(v.x) | ((unsigned)f2bf(v.y) << 16);
        o.y = (unsigned)f2bf(v.z) | ((unsigned)f2bf(v.w) << 16);
        *(U64*)(out + i) = o;
    }
}

// ---------------- batched transpose+convert: [E][R][Cc] f32 -> [E][Cc][R] bf16 ----------------
__global__ __launch_bounds__(256) void k_transpose_bf16(const float* __restrict__ in,
                                                        unsigned short* __restrict__ out,
                                                        int R, int Cc) {
    __shared__ unsigned short tile[32][33];
    long eoff = (long)blockIdx.z * R * Cc;
    int r0 = blockIdx.x * 32, c0 = blockIdx.y * 32;
    int tx = threadIdx.x & 31, ty = threadIdx.x >> 5;   // ty: 0..7
    #pragma unroll
    for (int i = 0; i < 32; i += 8)
        tile[ty + i][tx] = f2bf(in[eoff + (long)(r0 + ty + i) * Cc + c0 + tx]);
    __syncthreads();
    #pragma unroll
    for (int i = 0; i < 32; i += 8)
        out[eoff + (long)(c0 + ty + i) * R + r0 + tx] = tile[tx][ty + i];
}

// ---------------- init: zero counts, perm = -1 ----------------
__global__ void k_init(int* __restrict__ counts, int* __restrict__ perm, int cap) {
    int i = blockIdx.x * 256 + threadIdx.x;
    if (i < NROUTED) counts[i] = 0;
    if (i < cap) perm[i] = -1;
}

// ---------------- routing: one wave per token ----------------
__global__ __launch_bounds__(256) void k_route(const float* __restrict__ x,
                                               const float* __restrict__ gate_w,
                                               const float* __restrict__ lb_bias,
                                               int n, int* __restrict__ counts,
                                               int* __restrict__ r_e, int* __restrict__ r_ls,
                                               float* __restrict__ r_w) {
    int wid = threadIdx.x >> 5, lane = threadIdx.x & 31;
    int t = blockIdx.x * 8 + wid;
    if (t >= n) return;
    const float* xr = x + (long)t * CDIM;
    float acc[NROUTED];
    #pragma unroll
    for (int e = 0; e < NROUTED; ++e) acc[e] = 0.0f;
    for (int k = lane; k < CDIM; k += 32) {
        float xv = xr[k];
        #pragma unroll
        for (int e = 0; e < NROUTED; ++e) acc[e] += xv * gate_w[e * CDIM + k];
    }
    #pragma unroll
    for (int e = 0; e < NROUTED; ++e)
        for (int off = 16; off; off >>= 1) acc[e] += __shfl_xor(acc[e], off, 32);
    if (lane == 0) {
        float m = acc[0];
        #pragma unroll
        for (int e = 1; e < NROUTED; ++e) m = fmaxf(m, acc[e]);
        float p[NROUTED], s = 0.0f, b[NROUTED];
        #pragma unroll
        for (int e = 0; e < NROUTED; ++e) { p[e] = expf(acc[e] - m); s += p[e]; b[e] = acc[e] + lb_bias[e]; }
        float inv = 1.0f / s;
        int e0 = 0; float bv = b[0];
        #pragma unroll
        for (int e = 1; e < NROUTED; ++e) if (b[e] > bv) { bv = b[e]; e0 = e; }   // first max on ties
        int e1 = (e0 == 0) ? 1 : 0; float bv1 = b[e1];
        #pragma unroll
        for (int e = 0; e < NROUTED; ++e) if (e != e0 && b[e] > bv1) { bv1 = b[e]; e1 = e; }
        int sel[2] = { e0, e1 };
        #pragma unroll
        for (int j = 0; j < 2; ++j) {
            int e = sel[j];
            int ls = atomicAdd(&counts[e], 1);
            r_e[2 * t + j]  = e;
            r_ls[2 * t + j] = ls;
            r_w[2 * t + j]  = p[e] * inv;
        }
    }
}

// ---------------- 128-aligned offsets + tile->expert table ----------------
__global__ void k_offsets(const int* __restrict__ counts, int* __restrict__ offs,
                          int* __restrict__ tileExpert, int maxTiles) {
    int off = 0;
    for (int e = 0; e < NROUTED; ++e) {
        offs[e] = off;
        int nt = (counts[e] + BM - 1) / BM;
        for (int i = 0; i < nt; ++i) tileExpert[off / BM + i] = e;
        off += nt * BM;
    }
    offs[NROUTED] = off;
    for (int tt = off / BM; tt < maxTiles; ++tt) tileExpert[tt] = -1;
}

// ---------------- scatter tokens to global slots ----------------
__global__ void k_scatter(const int* __restrict__ r_e, const int* __restrict__ r_ls,
                          const int* __restrict__ offs, int* __restrict__ perm,
                          int* __restrict__ tokslot, int n) {
    int t = blockIdx.x * 256 + threadIdx.x;
    if (t >= n) return;
    #pragma unroll
    for (int j = 0; j < 2; ++j) {
        int e = r_e[2 * t + j];
        int g = offs[e] + r_ls[2 * t + j];
        perm[g] = t;
        tokslot[2 * t + j] = g;
    }
}

// ---------------- grouped bf16 WMMA GEMM: Y = act(A[rowidx] x W_e^T) ----------------
// A: [rowsA][K] bf16 row-major. W: [E][N][K] bf16 (N-major). Tile 128x128x32, 8 waves.
// Pad rows (rowidx < 0) are clamped to row 0: their garbage only lands in pad output
// rows, which the combine stage never reads.
__global__ __launch_bounds__(256) void k_moe_gemm_bf16(
    const unsigned short* __restrict__ A, const unsigned short* __restrict__ W,
    void* __restrict__ Y, int K, int N,
    const int* __restrict__ rowidx, const int* __restrict__ tileExpert,
    long wStride, int doGelu)
{
    __shared__ unsigned short As[BM * LDSK];
    __shared__ unsigned short Bs[BN * LDSK];

    int e = 0;
    if (tileExpert) { e = tileExpert[blockIdx.x]; if (e < 0) return; }
    const unsigned short* Wp = W + (long)e * wStride;
    const int rowBase = blockIdx.x * BM;
    const int colBase = blockIdx.y * BN;
    const int tid = threadIdx.x;

    // staging: thread owns chunks (row=srow,kc) and (row=srow+64,kc); 16B each
    const int srow = tid >> 2;
    const int skc  = tid & 3;
    int ga0 = rowBase + srow, ga1 = rowBase + srow + 64;
    int sa0 = rowidx ? rowidx[ga0] : ga0;
    int sa1 = rowidx ? rowidx[ga1] : ga1;
    if (sa0 < 0) sa0 = 0;   // pad row: any valid source, result row is unread
    if (sa1 < 0) sa1 = 0;
    const unsigned short* aPtr0 = A + (long)sa0 * K + skc * 8;
    const unsigned short* aPtr1 = A + (long)sa1 * K + skc * 8;
    const unsigned short* bPtr0 = Wp + (long)(colBase + srow) * K + skc * 8;
    const unsigned short* bPtr1 = Wp + (long)(colBase + srow + 64) * K + skc * 8;
    unsigned short* asw0 = &As[srow * LDSK + skc * 8];
    unsigned short* asw1 = &As[(srow + 64) * LDSK + skc * 8];
    unsigned short* bsw0 = &Bs[srow * LDSK + skc * 8];
    unsigned short* bsw1 = &Bs[(srow + 64) * LDSK + skc * 8];

    const int lane = tid & 31, wid = tid >> 5;
    const int waveM = wid & 3, waveN = wid >> 2;       // 4 M-waves x 2 N-waves
    const int h = lane >> 4, ln = lane & 15;

    // ISA layouts: A lane(m=ln): K = 8h+[0,8) then 16+8h+[0,8). B lane(n=ln): same K order.
    const unsigned short* afr[2];
    #pragma unroll
    for (int fm = 0; fm < 2; ++fm) afr[fm] = &As[(waveM * 32 + fm * 16 + ln) * LDSK + 8 * h];
    const unsigned short* bfr[4];
    #pragma unroll
    for (int fn = 0; fn < 4; ++fn) bfr[fn] = &Bs[(waveN * 64 + fn * 16 + ln) * LDSK + 8 * h];

    v8f acc[2][4] = {};

    U128 ra0 = *(const U128*)aPtr0;
    U128 ra1 = *(const U128*)aPtr1;
    U128 rb0 = *(const U128*)bPtr0;
    U128 rb1 = *(const U128*)bPtr1;

    for (int k0 = 0; k0 < K; k0 += BK) {
        __syncthreads();
        *(U128*)asw0 = ra0; *(U128*)asw1 = ra1;
        *(U128*)bsw0 = rb0; *(U128*)bsw1 = rb1;
        __syncthreads();
        int kn = k0 + BK;
        if (kn < K) {   // prefetch next tile into registers (overlaps WMMA below)
            ra0 = *(const U128*)(aPtr0 + kn);
            ra1 = *(const U128*)(aPtr1 + kn);
            rb0 = *(const U128*)(bPtr0 + kn);
            rb1 = *(const U128*)(bPtr1 + kn);
        }
        F256 fa[2], fb[4];
        #pragma unroll
        for (int fm = 0; fm < 2; ++fm) {
            fa[fm].a = *(const U128*)(afr[fm]);
            fa[fm].b = *(const U128*)(afr[fm] + 16);
        }
        #pragma unroll
        for (int fn = 0; fn < 4; ++fn) {
            fb[fn].a = *(const U128*)(bfr[fn]);
            fb[fn].b = *(const U128*)(bfr[fn] + 16);
        }
        #pragma unroll
        for (int fm = 0; fm < 2; ++fm) {
            v16bf va = __builtin_bit_cast(v16bf, fa[fm]);
            #pragma unroll
            for (int fn = 0; fn < 4; ++fn) {
                v16bf vb = __builtin_bit_cast(v16bf, fb[fn]);
                acc[fm][fn] = __builtin_amdgcn_wmma_f32_16x16x32_bf16(
                    false, va, false, vb, (short)0, acc[fm][fn], false, false);
            }
        }
    }

    // epilogue: C/D layout lane(n=ln), VGPR r -> M = 8h + r
    #pragma unroll
    for (int fm = 0; fm < 2; ++fm) {
        #pragma unroll
        for (int fn = 0; fn < 4; ++fn) {
            int grow = rowBase + waveM * 32 + fm * 16 + 8 * h;
            int gcol = colBase + waveN * 64 + fn * 16 + ln;
            if (doGelu) {
                unsigned short* O = (unsigned short*)Y;
                #pragma unroll
                for (int r = 0; r < 8; ++r)
                    O[(long)(grow + r) * N + gcol] = f2bf(gelu_exact(acc[fm][fn][r]));
            } else {
                float* O = (float*)Y;
                #pragma unroll
                for (int r = 0; r < 8; ++r)
                    O[(long)(grow + r) * N + gcol] = acc[fm][fn][r];
            }
        }
    }
}

// ---------------- combine: out[t] += w0*yr[g0] + w1*yr[g1] ----------------
__global__ __launch_bounds__(256) void k_combine(float* __restrict__ out, const float* __restrict__ yr,
                                                 const int* __restrict__ tokslot,
                                                 const float* __restrict__ r_w) {
    int t = blockIdx.x;
    int g0 = tokslot[2 * t], g1 = tokslot[2 * t + 1];
    float w0 = r_w[2 * t], w1 = r_w[2 * t + 1];
    int c = threadIdx.x * 4;
    float4 o  = *(float4*)(out + (long)t * CDIM + c);
    float4 y0 = *(const float4*)(yr + (long)g0 * CDIM + c);
    float4 y1 = *(const float4*)(yr + (long)g1 * CDIM + c);
    o.x += w0 * y0.x + w1 * y1.x;
    o.y += w0 * y0.y + w1 * y1.y;
    o.z += w0 * y0.z + w1 * y1.z;
    o.w += w0 * y0.w + w1 * y1.w;
    *(float4*)(out + (long)t * CDIM + c) = o;
}

extern "C" void kernel_launch(void* const* d_in, const int* in_sizes, int n_in,
                              void* d_out, int out_size, void* d_ws, size_t ws_size,
                              hipStream_t stream) {
    const float* x            = (const float*)d_in[0];
    const float* gate_w       = (const float*)d_in[1];
    const float* lb_bias      = (const float*)d_in[2];
    const float* shared_wfc   = (const float*)d_in[3];
    const float* shared_wproj = (const float*)d_in[4];
    const float* routed_wfc   = (const float*)d_in[5];
    const float* routed_wproj = (const float*)d_in[6];
    float* out = (float*)d_out;

    const int n = in_sizes[0] / CDIM;            // 8192 tokens
    const int maxTiles = (2 * n) / BM + NROUTED; // 135 worst case (padded segments)
    const int cap = maxTiles * BM;               // 17280 slots

    char* p = (char*)d_ws;
    auto carve = [&](size_t bytes) -> void* {
        void* r = (void*)p;
        p += (bytes + 255) & ~(size_t)255;
        return r;
    };
    unsigned short* x_bf       = (unsigned short*)carve((size_t)n * CDIM * 2);
    unsigned short* sh_wfc_t   = (unsigned short*)carve((size_t)HDIM * CDIM * 2);
    unsigned short* sh_wproj_t = (unsigned short*)carve((size_t)CDIM * HDIM * 2);
    unsigned short* r_wfc_t    = (unsigned short*)carve((size_t)NROUTED * HDIM * CDIM * 2);
    unsigned short* r_wproj_t  = (unsigned short*)carve((size_t)NROUTED * CDIM * HDIM * 2);
    unsigned short* hidden     = (unsigned short*)carve((size_t)cap * HDIM * 2);
    float*          yr         = (float*)carve((size_t)cap * CDIM * 4);
    int*   counts   = (int*)carve(64 * 4);
    int*   offs     = (int*)carve(64 * 4);
    int*   tileExpert = (int*)carve((size_t)maxTiles * 4);
    int*   perm     = (int*)carve((size_t)cap * 4);
    int*   r_e      = (int*)carve((size_t)2 * n * 4);
    int*   r_ls     = (int*)carve((size_t)2 * n * 4);
    float* r_w      = (float*)carve((size_t)2 * n * 4);
    int*   tokslot  = (int*)carve((size_t)2 * n * 4);

    // 1. x -> bf16
    long nx = (long)n * CDIM;
    k_cvt_bf16<<<(int)((nx / 4 + 255) / 256), 256, 0, stream>>>(x, x_bf, nx);
    // 2. weights -> bf16, transposed to [N][K]
    k_transpose_bf16<<<dim3(CDIM / 32, HDIM / 32, 1), 256, 0, stream>>>(shared_wfc, sh_wfc_t, CDIM, HDIM);
    k_transpose_bf16<<<dim3(HDIM / 32, CDIM / 32, 1), 256, 0, stream>>>(shared_wproj, sh_wproj_t, HDIM, CDIM);
    k_transpose_bf16<<<dim3(CDIM / 32, HDIM / 32, NROUTED), 256, 0, stream>>>(routed_wfc, r_wfc_t, CDIM, HDIM);
    k_transpose_bf16<<<dim3(HDIM / 32, CDIM / 32, NROUTED), 256, 0, stream>>>(routed_wproj, r_wproj_t, HDIM, CDIM);
    // 3. routing + dispatch tables
    k_init<<<(cap + 255) / 256, 256, 0, stream>>>(counts, perm, cap);
    k_route<<<(n + 7) / 8, 256, 0, stream>>>(x, gate_w, lb_bias, n, counts, r_e, r_ls, r_w);
    k_offsets<<<1, 1, 0, stream>>>(counts, offs, tileExpert, maxTiles);
    k_scatter<<<(n + 255) / 256, 256, 0, stream>>>(r_e, r_ls, offs, perm, tokslot, n);
    // 4. shared expert (dense, writes out)
    k_moe_gemm_bf16<<<dim3(n / BM, HDIM / BN), 256, 0, stream>>>(
        x_bf, sh_wfc_t, hidden, CDIM, HDIM, nullptr, nullptr, 0, 1);
    k_moe_gemm_bf16<<<dim3(n / BM, CDIM / BN), 256, 0, stream>>>(
        hidden, sh_wproj_t, out, HDIM, CDIM, nullptr, nullptr, 0, 0);
    // 5. routed experts (grouped, gathered via perm)
    k_moe_gemm_bf16<<<dim3(maxTiles, HDIM / BN), 256, 0, stream>>>(
        x_bf, r_wfc_t, hidden, CDIM, HDIM, perm, tileExpert, (long)CDIM * HDIM, 1);
    k_moe_gemm_bf16<<<dim3(maxTiles, CDIM / BN), 256, 0, stream>>>(
        hidden, r_wproj_t, yr, HDIM, CDIM, nullptr, tileExpert, (long)HDIM * CDIM, 0);
    // 6. weighted combine
    k_combine<<<n, 256, 0, stream>>>(out, yr, tokslot, r_w);
}
